// ExLlama_2946347565151
// MI455X (gfx1250) — matrix operand order repaired and checked by reading, not compiled, not run on hardware
//
#include <hip/hip_runtime.h>
#include <hip/hip_bf16.h>

typedef __attribute__((ext_vector_type(16))) _Float16 v16h;
typedef __attribute__((ext_vector_type(8)))  _Float16 v8h;
typedef __attribute__((ext_vector_type(8)))  float    v8f;
typedef unsigned int v4u __attribute__((ext_vector_type(4)));
typedef int v4i __attribute__((ext_vector_type(4)));
typedef int v8i __attribute__((ext_vector_type(8)));

#define S_     2048
#define HID_   4096
#define INTER_ 11008
#define HEADS_ 32
#define HD_    128
#define GS_    128

#if __has_builtin(__builtin_amdgcn_tensor_load_to_lds) && \
    __has_builtin(__builtin_amdgcn_s_wait_tensorcnt)
#define HAVE_TDM 1
#else
#define HAVE_TDM 0
#endif

#if HAVE_TDM
// ---------------------------------------------------------------------------
// TDM 2D tile load (f16 elements): global[tile_h][tile_w] (row stride in
// elements) -> contiguous LDS at lds_off. D# per CDNA5 ISA ch.8.
// Must be called with uniform (wave-scalar) arguments.
// ---------------------------------------------------------------------------
__device__ __forceinline__ void tdm_load_2d_f16(
    unsigned lds_off, unsigned long long gaddr,
    unsigned tile_w, unsigned tile_h, unsigned row_stride_elems) {
  v4u g0;
  g0[0] = 1u;                                          // count=1 valid D#
  g0[1] = lds_off;                                     // lds_addr (bytes)
  g0[2] = (unsigned)(gaddr & 0xFFFFFFFFu);             // global_addr[31:0]
  g0[3] = (unsigned)((gaddr >> 32) & 0x01FFFFFFu)      // global_addr[56:32]
          | (2u << 30);                                // type = 2 ("image")
  v8i g1;
  g1[0] = (int)(1u << 16);                             // data_size=1 (2B), mask=0
  g1[1] = (int)((tile_w & 0xFFFFu) << 16);             // tensor_dim0[15:0]
  g1[2] = (int)(((tile_w >> 16) & 0xFFFFu)             // tensor_dim0[31:16]
          | ((tile_h & 0xFFFFu) << 16));               // tensor_dim1[15:0]
  g1[3] = (int)(((tile_h >> 16) & 0xFFFFu)             // tensor_dim1[31:16]
          | ((tile_w & 0xFFFFu) << 16));               // tile_dim0
  g1[4] = (int)(tile_h & 0xFFFFu);                     // tile_dim1 (tile_dim2=0)
  g1[5] = (int)row_stride_elems;                       // tensor_dim0_stride lo
  g1[6] = 0;                                           // stride0 hi | stride1 lo
  g1[7] = 0;
  v4i g2 = {0, 0, 0, 0};
  v4i g3 = {0, 0, 0, 0};
#if __clang_major__ >= 23
  v8i g4 = {0, 0, 0, 0, 0, 0, 0, 0};
  __builtin_amdgcn_tensor_load_to_lds(g0, g1, g2, g3, g4, 0);
#else
  __builtin_amdgcn_tensor_load_to_lds(g0, g1, g2, g3, 0);
#endif
}
#endif

// ---------------------------------------------------------------------------
// RMSNorm: fp32 row in -> f16 row out. One block (256 thr = 8 waves) per row.
// ---------------------------------------------------------------------------
__global__ __launch_bounds__(256) void rmsnorm_f16(
    const float* __restrict__ x, const float* __restrict__ w,
    _Float16* __restrict__ out, int ncol) {
  int row = blockIdx.x;
  const float* xr = x + (size_t)row * ncol;
  __shared__ float red[8];
  float acc = 0.f;
  for (int i = threadIdx.x; i < ncol; i += 256) { float v = xr[i]; acc += v * v; }
  #pragma unroll
  for (int m = 16; m >= 1; m >>= 1) acc += __shfl_xor(acc, m, 32);
  if ((threadIdx.x & 31) == 0) red[threadIdx.x >> 5] = acc;
  __syncthreads();
  float tot = 0.f;
  #pragma unroll
  for (int i = 0; i < 8; ++i) tot += red[i];
  float inv = rsqrtf(tot / (float)ncol + 1e-6f);
  _Float16* orow = out + (size_t)row * ncol;
  for (int i = threadIdx.x; i < ncol; i += 256)
    orow[i] = (_Float16)(xr[i] * inv * w[i]);
}

// ---------------------------------------------------------------------------
// Fused GPTQ-int4 dequant + GEMM via v_wmma_f32_16x16x32_f16.
// Block = 256 thr = 8 waves as 2(M)x4(N) -> 32x64 tile. One 16x16 WMMA tile
// per wave, K-loop in steps of 32 (group-hoisted zero/scale).
// MODE 0: Y (f16) = X@W ; MODE 1: Yf (f32) = Res + X@W  (residual add)
// ---------------------------------------------------------------------------
template <int MODE>
__global__ __launch_bounds__(256) void gemm_w4(
    const _Float16* __restrict__ X,    // [M,K] f16 row-major
    const int* __restrict__ qw,        // [K/8, O]
    const int* __restrict__ qz,        // [K/GS, O/8]
    const float* __restrict__ sc,      // [K/GS, O]
    int K, int O,
    _Float16* __restrict__ Yh, const float* __restrict__ Res,
    float* __restrict__ Yf) {
  int tid  = threadIdx.x;
  int wave = tid >> 5, lane = tid & 31;
  int wm = wave >> 2, wn = wave & 3;
  int m0 = blockIdx.y * 32 + wm * 16;
  int n0 = blockIdx.x * 64 + wn * 16;
  int l15 = lane & 15, lh = lane >> 4;

  int mm   = m0 + l15;      // A: row per lane
  int kb8  = lh * 8;        // A: K sub-offset per lane-half
  int oo   = n0 + l15;      // B/D: output column per lane
  int kB16 = lh * 8;        // B: K base per lane-half

  v8f c = {};
  const int Od8 = O >> 3;
  const int nGroups = K / GS_;

  for (int g = 0; g < nGroups; ++g) {
    unsigned zq = (unsigned)qz[(size_t)g * Od8 + (oo >> 3)];
    float z     = (float)((zq >> ((oo & 7) * 4)) & 15u);
    float scale = sc[(size_t)g * O + oo];
    float zoff  = -(z + 1.0f) * scale;
    int kg = g * GS_;
    #pragma unroll
    for (int ks = 0; ks < GS_; ks += 32) {
      int kk = kg + ks;
      // ---- A fragment (f16 activations, two 16B contiguous chunks) ----
      const _Float16* xr = X + (size_t)mm * K + kk + kb8;
      v8h a0 = *(const v8h*)(xr);
      v8h a1 = *(const v8h*)(xr + 16);
      v16h a;
      #pragma unroll
      for (int e = 0; e < 8; ++e) { a[e] = a0[e]; a[8 + e] = a1[e]; }
      // ---- B fragment: dequant 16 int4 along K for this lane's column ----
      int r0 = (kk + kB16) >> 3;
      unsigned u0 = (unsigned)qw[(size_t)r0 * O + oo];
      unsigned u1 = (unsigned)qw[(size_t)(r0 + 2) * O + oo];
      __builtin_prefetch(&qw[(size_t)(r0 + 4) * O + oo], 0, 1);
      v16h b;
      #pragma unroll
      for (int e = 0; e < 8; ++e) {
        b[e]     = (_Float16)((float)((u0 >> (e * 4)) & 15u) * scale + zoff);
        b[8 + e] = (_Float16)((float)((u1 >> (e * 4)) & 15u) * scale + zoff);
      }
      c = __builtin_amdgcn_wmma_f32_16x16x32_f16(false, a, false, b,
                                                 (short)0, c, false, false);
    }
  }
  // ---- writeback per C/D layout: row = m0 + v + 8*lane_half ----
  #pragma unroll
  for (int v = 0; v < 8; ++v) {
    int row = m0 + v + 8 * lh;
    size_t idx = (size_t)row * O + oo;
    if (MODE == 0) Yh[idx] = (_Float16)c[v];
    else           Yf[idx] = Res[idx] + c[v];
  }
}

// ---------------------------------------------------------------------------
// RoPE in-place on Q and K (each thread owns the (d, d+64) rotation pair).
// ---------------------------------------------------------------------------
__global__ __launch_bounds__(256) void rope_qk(
    _Float16* __restrict__ q, _Float16* __restrict__ k,
    const float* __restrict__ sinT, const float* __restrict__ cosT) {
  int idx = blockIdx.x * 256 + threadIdx.x;     // over S*HID/2
  if (idx >= S_ * HID_ / 2) return;
  int s = idx / (HID_ / 2);
  int r = idx % (HID_ / 2);
  int head = r / (HD_ / 2);
  int d = r % (HD_ / 2);
  size_t base = (size_t)s * HID_ + head * HD_;
  float cs = cosT[s * HD_ + d], sn = sinT[s * HD_ + d];
  float q1 = (float)q[base + d], q2 = (float)q[base + d + HD_ / 2];
  q[base + d]           = (_Float16)(q1 * cs - q2 * sn);
  q[base + d + HD_ / 2] = (_Float16)(q2 * cs + q1 * sn);
  float k1 = (float)k[base + d], k2 = (float)k[base + d + HD_ / 2];
  k[base + d]           = (_Float16)(k1 * cs - k2 * sn);
  k[base + d + HD_ / 2] = (_Float16)(k2 * cs + k1 * sn);
}

// ---------------------------------------------------------------------------
// Flash attention. Block = 128 thr = 4 waves; block owns (head, 64 q-rows),
// each wave a 16-row q-tile. K tile staged to LDS via the Tensor Data Mover
// (wave 0 issues, TENSORcnt-tracked); V staged transposed by VALU so the
// P*V B-fragments are contiguous ds_load_b128. Online softmax uses shfl_xor
// row reductions (rows live in 16-lane halves of the WMMA C tile).
// ---------------------------------------------------------------------------
__global__ __launch_bounds__(128) void attn(
    const _Float16* __restrict__ Q, const _Float16* __restrict__ Km,
    const _Float16* __restrict__ Vm, _Float16* __restrict__ Obuf) {
  int head  = blockIdx.x;
  int qbase = blockIdx.y * 64;
  int tid = threadIdx.x, wave = tid >> 5, lane = tid & 31;
  int l15 = lane & 15, lh = lane >> 4;
  int qt0 = qbase + wave * 16;

  __shared__ _Float16 sK[32][HD_];        // 8 KB, key-major (TDM destination)
  __shared__ _Float16 sVt[HD_][32];       // 8 KB, d-major (transposed)
  __shared__ _Float16 sP[4][16][32];      // 4 KB, per-wave P tiles

  // Preload Q A-fragments for the 4 K(=HD) chunks of 32
  v16h aQ[4];
  {
    const _Float16* qr = Q + (size_t)(qt0 + l15) * HID_ + head * HD_;
    #pragma unroll
    for (int ch = 0; ch < 4; ++ch) {
      v8h a0 = *(const v8h*)(qr + ch * 32 + lh * 16);
      v8h a1 = *(const v8h*)(qr + ch * 32 + 8 + lh * 16);
      #pragma unroll
      for (int e = 0; e < 8; ++e) { aQ[ch][e] = a0[e]; aQ[ch][8 + e] = a1[e]; }
    }
  }
#if __has_builtin(__builtin_amdgcn_s_cluster_barrier)
  __builtin_amdgcn_s_cluster_barrier();   // NOP when not dispatched in cluster
#endif

  v8f zero = {};
  v8f oacc[8];
  #pragma unroll
  for (int j = 0; j < 8; ++j) oacc[j] = zero;
  float rmax[8], rsum[8];
  #pragma unroll
  for (int v = 0; v < 8; ++v) { rmax[v] = -1e30f; rsum[v] = 0.f; }
  const float scl = 0.08838834764831845f;  // 1/sqrt(128)

  for (int kk = 0; kk < qbase + 64; kk += 32) {   // uniform over block
    __syncthreads();
#if HAVE_TDM
    if (wave == 0) {
      // LDS aperture puts the byte offset in addr[31:0] of the generic ptr.
      unsigned ldsK = (unsigned)(size_t)(&sK[0][0]);
      unsigned long long ga =
          (unsigned long long)(size_t)(Km + (size_t)kk * HID_ + head * HD_);
      tdm_load_2d_f16(ldsK, ga, HD_, 32, HID_);   // 32x128 f16 tile
    }
#endif
    // ---- V staged transposed (and K too, if no TDM) ----
    #pragma unroll
    for (int u = 0; u < 4; ++u) {
      int unit = u * 128 + tid;                   // 512 v8h units total
      int row = unit >> 4, c8 = unit & 15;
#if !HAVE_TDM
      v8h dk = *(const v8h*)(Km + (size_t)(kk + row) * HID_ + head * HD_ + c8 * 8);
      *(v8h*)(&sK[row][c8 * 8]) = dk;
#endif
      v8h dv = *(const v8h*)(Vm + (size_t)(kk + row) * HID_ + head * HD_ + c8 * 8);
      #pragma unroll
      for (int e = 0; e < 8; ++e) sVt[c8 * 8 + e][row] = dv[e];
    }
#if HAVE_TDM
    if (wave == 0) __builtin_amdgcn_s_wait_tensorcnt((short)0);
#endif
    __syncthreads();

    if (kk <= qt0 + 15) {                         // wave has live keys here
      v8f st[2];
      float newmax[8];
      #pragma unroll
      for (int v = 0; v < 8; ++v) newmax[v] = rmax[v];

      #pragma unroll
      for (int t = 0; t < 2; ++t) {
        st[t] = zero;
        int kt = kk + t * 16;
        if (kt <= qt0 + 15) {
          #pragma unroll
          for (int ch = 0; ch < 4; ++ch) {
            v16h b = *(const v16h*)(&sK[t * 16 + l15][ch * 32 + lh * 16]);
            st[t] = __builtin_amdgcn_wmma_f32_16x16x32_f16(
                false, aQ[ch], false, b, (short)0, st[t], false, false);
          }
        }
        #pragma unroll
        for (int v = 0; v < 8; ++v) {             // scale + causal mask
          int qi = qt0 + v + 8 * lh;
          int ki = kt + l15;
          float s = st[t][v] * scl;
          if (ki > qi || kt > qt0 + 15) s = -1e30f;
          st[t][v] = s;
        }
      }
      // ---- online softmax ----
      #pragma unroll
      for (int v = 0; v < 8; ++v) {
        float m = fmaxf(st[0][v], st[1][v]);
        #pragma unroll
        for (int msk = 1; msk < 16; msk <<= 1) m = fmaxf(m, __shfl_xor(m, msk, 32));
        newmax[v] = fmaxf(newmax[v], m);
      }
      #pragma unroll
      for (int v = 0; v < 8; ++v) {
        float f  = __expf(rmax[v] - newmax[v]);
        float p0 = __expf(st[0][v] - newmax[v]);
        float p1 = __expf(st[1][v] - newmax[v]);
        float ps = p0 + p1;
        #pragma unroll
        for (int msk = 1; msk < 16; msk <<= 1) ps += __shfl_xor(ps, msk, 32);
        rsum[v] = rsum[v] * f + ps;
        rmax[v] = newmax[v];
        int m = v + 8 * lh;                       // C-layout -> LDS (f16)
        sP[wave][m][l15]      = (_Float16)p0;
        sP[wave][m][16 + l15] = (_Float16)p1;
        #pragma unroll
        for (int j = 0; j < 8; ++j) oacc[j][v] *= f;
      }
      asm volatile("s_wait_dscnt 0" ::: "memory"); // LDS store->load, same wave
      // ---- reload P in A-fragment layout ----
      v16h aP;
      {
        v8h p0 = *(const v8h*)(&sP[wave][l15][lh * 16]);
        v8h p1 = *(const v8h*)(&sP[wave][l15][8 + lh * 16]);
        #pragma unroll
        for (int e = 0; e < 8; ++e) { aP[e] = p0[e]; aP[8 + e] = p1[e]; }
      }
      // ---- O += P * V over 8 d-tiles ----
      #pragma unroll
      for (int j = 0; j < 8; ++j) {
        v16h b = *(const v16h*)(&sVt[j * 16 + l15][lh * 16]);
        oacc[j] = __builtin_amdgcn_wmma_f32_16x16x32_f16(
            false, aP, false, b, (short)0, oacc[j], false, false);
      }
    }
  }
  // ---- normalize + writeback ----
  #pragma unroll
  for (int v = 0; v < 8; ++v) {
    float inv = 1.0f / rsum[v];
    int row = qt0 + v + 8 * lh;
    _Float16* orow = Obuf + (size_t)row * HID_ + head * HD_;
    #pragma unroll
    for (int j = 0; j < 8; ++j)
      orow[j * 16 + l15] = (_Float16)(oacc[j][v] * inv);
  }
}

// ---------------------------------------------------------------------------
// y = silu(gate) * up  (in-place over gate buffer)
// ---------------------------------------------------------------------------
__global__ __launch_bounds__(256) void silu_mul(
    const _Float16* __restrict__ g, const _Float16* __restrict__ u,
    _Float16* __restrict__ y, long n) {
  long i = (long)blockIdx.x * 256 + threadIdx.x;
  if (i >= n) return;
  float gv = (float)g[i];
  float s = gv / (1.f + __expf(-gv));
  y[i] = (_Float16)(s * (float)u[i]);
}

// ---------------------------------------------------------------------------
extern "C" void kernel_launch(void* const* d_in, const int* in_sizes, int n_in,
                              void* d_out, int out_size, void* d_ws, size_t ws_size,
                              hipStream_t stream) {
  (void)in_sizes; (void)n_in; (void)out_size; (void)ws_size;
  // inputs: per name in {q,k,v,o,gate,up,down}: qw, qz, sc  -> idx 0..20
  const int*   qw_q = (const int*)d_in[0];  const int* qz_q = (const int*)d_in[1];  const float* sc_q = (const float*)d_in[2];
  const int*   qw_k = (const int*)d_in[3];  const int* qz_k = (const int*)d_in[4];  const float* sc_k = (const float*)d_in[5];
  const int*   qw_v = (const int*)d_in[6];  const int* qz_v = (const int*)d_in[7];  const float* sc_v = (const float*)d_in[8];
  const int*   qw_o = (const int*)d_in[9];  const int* qz_o = (const int*)d_in[10]; const float* sc_o = (const float*)d_in[11];
  const int*   qw_g = (const int*)d_in[12]; const int* qz_g = (const int*)d_in[13]; const float* sc_g = (const float*)d_in[14];
  const int*   qw_u = (const int*)d_in[15]; const int* qz_u = (const int*)d_in[16]; const float* sc_u = (const float*)d_in[17];
  const int*   qw_d = (const int*)d_in[18]; const int* qz_d = (const int*)d_in[19]; const float* sc_d = (const float*)d_in[20];
  const float* hidden = (const float*)d_in[21];
  const float* ln1    = (const float*)d_in[22];
  const float* ln2    = (const float*)d_in[23];
  const float* sinT   = (const float*)d_in[24];
  const float* cosT   = (const float*)d_in[25];
  float* out = (float*)d_out;

  // workspace layout
  char* ws = (char*)d_ws;
  const size_t F32ROWS = (size_t)S_ * HID_ * sizeof(float);     // 32 MB
  const size_t F16ROWS = (size_t)S_ * HID_ * sizeof(_Float16);  // 16 MB
  float*    hbuf  = (float*)ws;
  _Float16* xn    = (_Float16*)(ws + F32ROWS);
  _Float16* qb    = (_Float16*)(ws + F32ROWS + F16ROWS * 1);
  _Float16* kb    = (_Float16*)(ws + F32ROWS + F16ROWS * 2);
  _Float16* vb    = (_Float16*)(ws + F32ROWS + F16ROWS * 3);
  _Float16* ob    = (_Float16*)(ws + F32ROWS + F16ROWS * 4);
  _Float16* gateb = (_Float16*)(ws + F32ROWS + F16ROWS * 5);
  _Float16* upb   = gateb + (size_t)S_ * INTER_;

  dim3 blk(256);
  dim3 gHID(HID_ / 64, S_ / 32);
  dim3 gINT(INTER_ / 64, S_ / 32);

  // 1) x = rms(h, ln1)
  rmsnorm_f16<<<S_, blk, 0, stream>>>(hidden, ln1, xn, HID_);
  // 2) q/k/v projections (int4 dequant + WMMA GEMM)
  gemm_w4<0><<<gHID, blk, 0, stream>>>(xn, qw_q, qz_q, sc_q, HID_, HID_, qb, nullptr, nullptr);
  gemm_w4<0><<<gHID, blk, 0, stream>>>(xn, qw_k, qz_k, sc_k, HID_, HID_, kb, nullptr, nullptr);
  gemm_w4<0><<<gHID, blk, 0, stream>>>(xn, qw_v, qz_v, sc_v, HID_, HID_, vb, nullptr, nullptr);
  // 3) RoPE on q,k
  rope_qk<<<(S_ * HID_ / 2 + 255) / 256, blk, 0, stream>>>(qb, kb, sinT, cosT);
  // 4) flash attention (TDM-staged K tiles)
  attn<<<dim3(HEADS_, S_ / 64), 128, 0, stream>>>(qb, kb, vb, ob);
  // 5) h = hidden + o @ Wo
  gemm_w4<1><<<gHID, blk, 0, stream>>>(ob, qw_o, qz_o, sc_o, HID_, HID_, nullptr, hidden, hbuf);
  // 6) x = rms(h, ln2)
  rmsnorm_f16<<<S_, blk, 0, stream>>>(hbuf, ln2, xn, HID_);
  // 7) gate / up projections
  gemm_w4<0><<<gINT, blk, 0, stream>>>(xn, qw_g, qz_g, sc_g, HID_, INTER_, gateb, nullptr, nullptr);
  gemm_w4<0><<<gINT, blk, 0, stream>>>(xn, qw_u, qz_u, sc_u, HID_, INTER_, upb, nullptr, nullptr);
  // 8) y = silu(gate) * up
  silu_mul<<<((long)S_ * INTER_ + 255) / 256, blk, 0, stream>>>(gateb, upb, gateb, (long)S_ * INTER_);
  // 9) out = h + y @ Wdown
  gemm_w4<1><<<gHID, blk, 0, stream>>>(gateb, qw_d, qz_d, sc_d, INTER_, HID_, nullptr, hbuf, out);
}
